// PFNetDense_8778913153238
// MI455X (gfx1250) — compile-verified, matrix operand order checked by reading, not run on hardware
//
#include <hip/hip_runtime.h>
#include <hip/hip_bf16.h>

// ---------------------------------------------------------------------------
// Problem constants (from reference)
// ---------------------------------------------------------------------------
#define Bb     8
#define Nn     12800
#define Ff     256       // feature dim
#define Dd     128       // distance-feature dim
#define NBINS  100       // N / BIN_SIZE
#define BS     128       // bin size
#define Pp     (Bb * Nn) // 102400 total points
#define NKEY   256       // key range for counting sort (bin_idx <= 198)

// ---------------------------------------------------------------------------
// WMMA types & helpers (CDNA5 gfx1250, wave32)
// ---------------------------------------------------------------------------
typedef __bf16 v16bf __attribute__((ext_vector_type(16)));
typedef float  v8f   __attribute__((ext_vector_type(8)));
typedef unsigned int u32x4 __attribute__((ext_vector_type(4)));
typedef unsigned int u32x8 __attribute__((ext_vector_type(8)));

union FragU { v16bf v; uint4 q[2]; unsigned short h[16]; };
union Pack8 { unsigned short h[8]; uint4 q; };

__device__ __forceinline__ unsigned short f2bf(float f) {
    unsigned int u = __float_as_uint(f);
    unsigned int r = u + 0x7fffu + ((u >> 16) & 1u); // round-to-nearest-even
    return (unsigned short)(r >> 16);
}
__device__ __forceinline__ float bf2f(unsigned short h) {
    return __uint_as_float(((unsigned int)h) << 16);
}

// Load a 16x32 (A) or 32x16 (B, stored transposed/col-major) bf16 fragment.
// Layout per ISA 7.12.2: lanes 0-15 half=0, lanes 16-31 half=1;
//   VGPR0..3 = K {half*8 .. half*8+7}, VGPR4..7 = K {16+half*8 ..}.
__device__ __forceinline__ v16bf ldfrag(const unsigned short* base, int row,
                                        int stride, int k0, int lane) {
    const unsigned short* p =
        base + (size_t)row * stride + k0 + ((lane >> 4) << 3);
    FragU f;
    f.q[0] = *(const uint4*)(p);
    f.q[1] = *(const uint4*)(p + 16);
    return f.v;
}

__device__ __forceinline__ v16bf zfrag() {
    FragU f;
    f.q[0] = make_uint4(0u, 0u, 0u, 0u);
    f.q[1] = f.q[0];
    return f.v;
}

__device__ __forceinline__ v8f wmma_bf16(v16bf a, v16bf b, v8f c) {
    return __builtin_amdgcn_wmma_f32_16x16x32_bf16(
        false, a, false, b, (short)0, c, false, false);
}

__device__ __forceinline__ float eluf(float x) {
    return x > 0.f ? x : (__expf(x) - 1.f);
}

// ---------------------------------------------------------------------------
// CDNA5 async global->LDS copy (per-lane, ASYNCcnt-tracked) + TDM 1-D tile.
// ---------------------------------------------------------------------------
__device__ __forceinline__ void async_copy_b128(unsigned lds_off,
                                                const void* gaddr) {
    asm volatile("global_load_async_to_lds_b128 %0, %1, off"
                 :: "v"(lds_off), "v"(gaddr)
                 : "memory");
}
__device__ __forceinline__ void wait_asynccnt0() {
    asm volatile("s_wait_asynccnt 0x0" ::: "memory");
}

// 1-D tensor DMA: copy `nelems` bf16 elements from contiguous global memory
// to LDS at byte offset `lds_off`. Descriptor per ISA 8.3/8.4 (D# group0/1):
//  g0: count=1 | lds_addr | global_addr[56:0] | type=2
//  g1: data_size=1 (2B), tensor_dim0 = nelems, tile_dim0 = nelems (1 row)
__device__ __forceinline__ void tdm_load_1d(unsigned lds_off, const void* gptr,
                                            unsigned nelems) {
    unsigned long long ga = (unsigned long long)gptr;
    u32x4 g0;
    g0[0] = 1u;                                       // count=1, user mode
    g0[1] = lds_off;                                  // lds_addr [63:32]
    g0[2] = (unsigned)(ga & 0xffffffffu);             // global_addr [95:64]
    g0[3] = ((unsigned)(ga >> 32) & 0x01ffffffu) |    // global_addr [120:96]
            (2u << 30);                               // type=2 (image)
    u32x8 g1;
    g1[0] = (1u << 16);                               // data_size=1 -> 2 bytes
    g1[1] = (nelems & 0xffffu) << 16;                 // tensor_dim0 [63:48]
    g1[2] = (nelems >> 16) & 0xffffu;                 // tensor_dim0 [79:64]
    g1[3] = (nelems & 0xffffu) << 16;                 // tile_dim0  [127:112]
    g1[4] = 0u; g1[5] = 0u; g1[6] = 0u; g1[7] = 0u;   // dims 1/2 unused
    u32x4 gz; gz[0] = 0u; gz[1] = 0u; gz[2] = 0u; gz[3] = 0u;
    asm volatile("tensor_load_to_lds %0, %1, %2, %3"
                 :: "s"(g0), "s"(g1), "s"(gz), "s"(gz)
                 : "memory");
}

// ---------------------------------------------------------------------------
// K0: transpose + fp32 -> bf16 weight conversion:  w[k][n] -> wt[n][k]
// ---------------------------------------------------------------------------
__global__ void k_transpose_bf16(const float* __restrict__ w,
                                 unsigned short* __restrict__ wt,
                                 int rows, int cols) {
    int i = blockIdx.x * blockDim.x + threadIdx.x;
    if (i < rows * cols) {
        int k = i / cols, n = i % cols;
        wt[(size_t)n * rows + k] = f2bf(w[i]);
    }
}

// ---------------------------------------------------------------------------
// K1: LayerNorm over F=256, one wave per point, output bf16
// ---------------------------------------------------------------------------
__global__ void __launch_bounds__(256) k_layernorm(
    const float* __restrict__ x, const float* __restrict__ g,
    const float* __restrict__ be, unsigned short* __restrict__ xn) {
    int p    = (blockIdx.x * blockDim.x + threadIdx.x) >> 5;
    int lane = threadIdx.x & 31;
    const float* xp = x + (size_t)p * Ff + lane * 8;
    float4 q0 = *(const float4*)xp;
    float4 q1 = *(const float4*)(xp + 4);
    float v[8] = {q0.x, q0.y, q0.z, q0.w, q1.x, q1.y, q1.z, q1.w};
    float s = 0.f, sq = 0.f;
#pragma unroll
    for (int j = 0; j < 8; ++j) { s += v[j]; sq += v[j] * v[j]; }
#pragma unroll
    for (int o = 16; o > 0; o >>= 1) {
        s  += __shfl_xor(s, o, 32);
        sq += __shfl_xor(sq, o, 32);
    }
    float mu  = s * (1.f / Ff);
    float var = sq * (1.f / Ff) - mu * mu;
    float rs  = rsqrtf(var + 1e-6f);
    Pack8 pk;
#pragma unroll
    for (int j = 0; j < 8; ++j) {
        float y = (v[j] - mu) * rs * g[lane * 8 + j] + be[lane * 8 + j];
        pk.h[j] = f2bf(y);
    }
    *(uint4*)(xn + (size_t)p * Ff + lane * 8) = pk.q;
}

// ---------------------------------------------------------------------------
// K2/K3: bf16 WMMA GEMM  O[P x Ncols] = act(A[P x K] * BT^T + bias)
// A tile (128 rows x K, contiguous) and full BT (Ncols x K, contiguous) are
// staged into LDS with two 1-D TDM loads; all fragment reads hit LDS.
// ---------------------------------------------------------------------------
__global__ void __launch_bounds__(256) k_gemm_bias_act(
    const unsigned short* __restrict__ A, const unsigned short* __restrict__ BT,
    const float* __restrict__ bias, unsigned short* __restrict__ Obf,
    float* __restrict__ Of32, int K, int Ncols, int act) {
    extern __shared__ unsigned char gsm[];
    unsigned short* s_a = (unsigned short*)gsm;   // [128][K]
    unsigned short* s_b = s_a + 128 * K;          // [Ncols][K]

    int wave = threadIdx.x >> 5, lane = threadIdx.x & 31;
    int rowbase = blockIdx.x * 128;

    if (wave == 0) {
        tdm_load_1d(0u, A + (size_t)rowbase * K, (unsigned)(128 * K));
        tdm_load_1d((unsigned)(128 * K * 2), BT, (unsigned)(Ncols * K));
        __builtin_amdgcn_s_wait_tensorcnt(0);
    }
    __syncthreads();

    int lrow   = wave * 16 + (lane & 15);
    int ntiles = Ncols >> 4, ktiles = K >> 5;
    for (int nt = 0; nt < ntiles; ++nt) {
        int bcol = nt * 16 + (lane & 15);
        v8f acc = {};
        for (int kt = 0; kt < ktiles; ++kt) {
            v16bf a = ldfrag(s_a, lrow, K, kt * 32, lane);
            v16bf b = ldfrag(s_b, bcol, K, kt * 32, lane);
            acc = wmma_bf16(a, b, acc);
        }
        int col = nt * 16 + (lane & 15);
        int mb  = (lane >> 4) * 8;
        float bv = bias[col];
#pragma unroll
        for (int r = 0; r < 8; ++r) {
            float o = acc[r] + bv;
            if (act) o = eluf(o);
            size_t idx = (size_t)(rowbase + wave * 16 + mb + r) * Ncols + col;
            if (Obf)  Obf[idx]  = f2bf(o);
            if (Of32) Of32[idx] = o;
        }
    }
}

// ---------------------------------------------------------------------------
// K4: LSH binning: mul = xd @ rot[:, :50]; argmax over [mul, -mul]; +mask off
// ---------------------------------------------------------------------------
__global__ void __launch_bounds__(128) k_bins(
    const float* __restrict__ xd, const float* __restrict__ rot,
    const unsigned char* __restrict__ msk, int* __restrict__ key) {
    __shared__ float s_rot[Dd * 50];
    for (int i = threadIdx.x; i < Dd * 50; i += blockDim.x) {
        int k = i / 50, j = i % 50;
        s_rot[i] = rot[k * 100 + j]; // rot row stride = MAXB/2 = 100
    }
    __syncthreads();
    int p = blockIdx.x * blockDim.x + threadIdx.x;
    const float* xp = xd + (size_t)p * Dd;
    float acc[50];
#pragma unroll
    for (int j = 0; j < 50; ++j) acc[j] = 0.f;
    for (int k = 0; k < Dd; ++k) {
        float xv = xp[k];
        const float* rr = &s_rot[k * 50];
#pragma unroll
        for (int j = 0; j < 50; ++j) acc[j] += xv * rr[j];
    }
    float best = acc[0]; int bi = 0;
#pragma unroll
    for (int j = 1; j < 50; ++j)
        if (acc[j] > best) { best = acc[j]; bi = j; }
#pragma unroll
    for (int j = 0; j < 50; ++j)
        if (-acc[j] > best) { best = -acc[j]; bi = j + 50; }
    int b = p / Nn, n = p % Nn;
    int k_ = bi + (msk[(size_t)b * Nn + n] ? 0 : (NBINS - 1));
    key[p] = k_;
}

// ---------------------------------------------------------------------------
// K5-K7: deterministic (stable) counting sort of keys -> flat permutation
// ---------------------------------------------------------------------------
__global__ void __launch_bounds__(256) k_histb(const int* __restrict__ key,
                                               int* __restrict__ bcnt) {
    __shared__ int hc[NKEY];
    int t = threadIdx.x;
    hc[t] = 0;
    __syncthreads();
    atomicAdd(&hc[key[blockIdx.x * 256 + t]], 1);
    __syncthreads();
    bcnt[(size_t)blockIdx.x * NKEY + t] = hc[t];
}

__global__ void k_scan(const int* __restrict__ bcnt, int* __restrict__ boff) {
    if (threadIdx.x != 0) return;
    int bb = blockIdx.x;                   // batch
    const int BLKS = Nn / 256;             // 50 blocks per batch
    int run = 0;
    for (int k = 0; k < NKEY; ++k)
        for (int blk = 0; blk < BLKS; ++blk) {
            size_t g = (size_t)(bb * BLKS + blk) * NKEY + k;
            boff[g] = run;
            run += bcnt[g];
        }
}

__global__ void __launch_bounds__(256) k_place(const int* __restrict__ key,
                                               const int* __restrict__ boff,
                                               int* __restrict__ flat) {
    __shared__ int sk[256];
    int t = threadIdx.x;
    int p = blockIdx.x * 256 + t;
    sk[t] = key[p];
    __syncthreads();
    int k = sk[t], rank = 0;
    for (int i = 0; i < 256; ++i) rank += (i < t && sk[i] == k) ? 1 : 0;
    int pos = boff[(size_t)blockIdx.x * NKEY + k] + rank;
    int b = p / Nn, n = p % Nn;
    flat[(size_t)b * Nn + pos] = n;
}

// ---------------------------------------------------------------------------
// K8: per-bin fused kernel: Gram -> adjacency -> 2x GHConv (all WMMA) -> out
// 1 workgroup per (batch, bin); 256 threads = 8 waves; dynamic LDS ~258 KB.
// Row gathers use GLOBAL_LOAD_ASYNC_TO_LDS_B128 (ASYNCcnt).
// ---------------------------------------------------------------------------
__global__ void __launch_bounds__(256, 1) k_binproc(
    const unsigned short* __restrict__ xn_bf,
    const unsigned short* __restrict__ xd_bf,
    const float* __restrict__ xd_f32, const int* __restrict__ flat,
    const unsigned char* __restrict__ msk,
    const unsigned short* __restrict__ thT0, const unsigned short* __restrict__ whT0,
    const unsigned short* __restrict__ wtT0, const float* __restrict__ bt0,
    const unsigned short* __restrict__ thT1, const unsigned short* __restrict__ whT1,
    const unsigned short* __restrict__ wtT1, const float* __restrict__ bt1,
    float* __restrict__ out) {
    extern __shared__ unsigned char smem[];
    unsigned short* s_xd  = (unsigned short*)smem;       // [128][128] gathered xd
    unsigned short* s_adj = s_xd + BS * Dd;              // [128][128] adjacency
    unsigned short* s_x   = s_adj + BS * BS;             // [128][256] activations
    unsigned short* s_ht  = s_x + BS * Ff;               // [256][128] H transposed
    unsigned short* s_y   = s_ht + Ff * BS;              // [128][256] layer output
    float* s_na   = (float*)(s_y + BS * Ff);             // [128]
    float* s_m    = s_na + BS;                           // [128]
    float* s_norm = s_m + BS;                            // [128]
    int*   s_idx  = (int*)(s_norm + BS);                 // [128]

    int b = blockIdx.x / NBINS, bin = blockIdx.x % NBINS;
    int t = threadIdx.x, wave = t >> 5, lane = t & 31;

    // ---- L2 warm-up for streamed weights (global_prefetch) ----
    __builtin_prefetch(thT0 + (size_t)t * Ff, 0, 2);
    __builtin_prefetch(whT0 + (size_t)t * Ff, 0, 2);
    __builtin_prefetch(wtT0 + (size_t)t * Ff, 0, 2);

    // ---- gather indices + mask ----
    if (t < BS) {
        int id   = flat[(size_t)b * Nn + bin * BS + t];
        s_idx[t] = id;
        s_m[t]   = msk[(size_t)b * Nn + id] ? 1.f : 0.f;
    }
    __syncthreads();

    // ---- async gather xd (bf16) and x (bf16) tiles into LDS ----
    for (int e = t; e < BS * Dd / 8; e += 256) {
        int row = e >> 4, c = (e & 15) << 3;
        const void* src = xd_bf + ((size_t)b * Nn + s_idx[row]) * Dd + c;
        unsigned dst = (unsigned)((const unsigned char*)&s_xd[row * Dd + c] -
                                  (const unsigned char*)smem);
        async_copy_b128(dst, src);
    }
    for (int e = t; e < BS * Ff / 8; e += 256) {
        int row = e >> 5, c = (e & 31) << 3;
        const void* src = xn_bf + ((size_t)b * Nn + s_idx[row]) * Ff + c;
        unsigned dst = (unsigned)((const unsigned char*)&s_x[row * Ff + c] -
                                  (const unsigned char*)smem);
        async_copy_b128(dst, src);
    }
    // ---- squared norms from f32 xd (overlaps with async gather) ----
    if (t < BS) {
        const float* xr = xd_f32 + ((size_t)b * Nn + s_idx[t]) * Dd;
        float s = 0.f;
        for (int k = 0; k < Dd; k += 4) {
            float4 q = *(const float4*)(xr + k);
            s += q.x * q.x + q.y * q.y + q.z * q.z + q.w * q.w;
        }
        s_na[t] = s;
    }
    wait_asynccnt0();
    __syncthreads();

    // ---- Gram matrix G = xd_b @ xd_b^T via WMMA; adjacency in LDS ----
    int mrow = wave * 16;
    for (int nt = 0; nt < 8; ++nt) {
        v8f acc = {};
        for (int kt = 0; kt < 4; ++kt) {
            v16bf a  = ldfrag(s_xd, mrow + (lane & 15), Dd, kt * 32, lane);
            v16bf bb = ldfrag(s_xd, nt * 16 + (lane & 15), Dd, kt * 32, lane);
            acc = wmma_bf16(a, bb, acc);
        }
        int col = nt * 16 + (lane & 15);
        int mb  = (lane >> 4) * 8;
#pragma unroll
        for (int r = 0; r < 8; ++r) {
            int row = mrow + mb + r;
            float d2 = s_na[row] + s_na[col] - 2.f * acc[r];
            float d  = sqrtf(fmaxf(d2, 1e-6f));
            float dm = __expf(-0.1f * d);
            dm = fminf(fmaxf(dm, 0.f), 1.f) * s_m[row] * s_m[col];
            s_adj[row * BS + col] = f2bf(dm);
        }
    }
    __syncthreads();

    // ---- degree / norm (shared by both layers; adj is fixed) ----
    if (t < BS) {
        float s = 0.f;
        for (int j = 0; j < BS; ++j) s += bf2f(s_adj[t * BS + j]);
        s = fminf(s, 1000.f);
        s_norm[t] = rsqrtf(s + 1e-6f) * s_m[t];
    }
    __syncthreads();

    // ---- two GHConv layers ----
    const unsigned short* thT[2] = {thT0, thT1};
    const unsigned short* whT[2] = {whT0, whT1};
    const unsigned short* wtT[2] = {wtT0, wtT1};
    const float* btp[2] = {bt0, bt1};
    unsigned short* cur = s_x;
    unsigned short* nxt = s_y;

    for (int layer = 0; layer < 2; ++layer) {
        // Phase A: H = ((x*m) @ theta) * norm, stored transposed (Ht[n][m])
        for (int nt = 0; nt < 16; ++nt) {
            int arow = mrow + (lane & 15);
            int bcol = nt * 16 + (lane & 15);
            float mrowv = s_m[arow];
            v8f acc = {};
            for (int kt = 0; kt < 8; ++kt) {
                v16bf a = ldfrag(cur, arow, Ff, kt * 32, lane);
                if (mrowv == 0.f) a = zfrag(); // mask == {0,1}: data select
                v16bf bb = ldfrag(thT[layer], bcol, Ff, kt * 32, lane);
                acc = wmma_bf16(a, bb, acc);
            }
            int col = nt * 16 + (lane & 15);
            int mb  = (lane >> 4) * 8;
            Pack8 pk;
#pragma unroll
            for (int r = 0; r < 8; ++r)
                pk.h[r] = f2bf(acc[r] * s_norm[mrow + mb + r]);
            *(uint4*)&s_ht[col * BS + mrow + mb] = pk.q;
        }
        __syncthreads();

        // Phase B: hom = adj @ H, het = (x*m)@wh, gate = sigmoid(x@wt+bt)
        for (int nt = 0; nt < 16; ++nt) {
            int arow = mrow + (lane & 15);
            int bcol = nt * 16 + (lane & 15);
            float mrowv = s_m[arow];
            v8f hom = {}, het = {}, gat = {};
            for (int kt = 0; kt < 4; ++kt) {
                v16bf a  = ldfrag(s_adj, arow, BS, kt * 32, lane);
                v16bf bb = ldfrag(s_ht, bcol, BS, kt * 32, lane);
                hom = wmma_bf16(a, bb, hom);
            }
            for (int kt = 0; kt < 8; ++kt) {
                v16bf a  = ldfrag(cur, arow, Ff, kt * 32, lane);
                v16bf am = a;
                if (mrowv == 0.f) am = zfrag();
                v16bf bh = ldfrag(whT[layer], bcol, Ff, kt * 32, lane);
                het = wmma_bf16(am, bh, het);
                v16bf bt_ = ldfrag(wtT[layer], bcol, Ff, kt * 32, lane);
                gat = wmma_bf16(a, bt_, gat);
            }
            int col = nt * 16 + (lane & 15);
            int mb  = (lane >> 4) * 8;
            float bias = btp[layer][col];
#pragma unroll
            for (int r = 0; r < 8; ++r) {
                int row = mrow + mb + r;
                float fh = hom[r] * s_norm[row];
                float g  = 1.f / (1.f + __expf(-(gat[r] + bias)));
                float o  = g * fh + (1.f - g) * het[r];
                o = eluf(o) * s_m[row];
                if (layer == 1)
                    out[((size_t)b * Nn + s_idx[row]) * Ff + col] = o;
                else
                    nxt[row * Ff + col] = f2bf(o);
            }
        }
        __syncthreads();
        unsigned short* tp = cur; cur = nxt; nxt = tp;
    }
}

// ---------------------------------------------------------------------------
// Host launcher
// ---------------------------------------------------------------------------
extern "C" void kernel_launch(void* const* d_in, const int* in_sizes, int n_in,
                              void* d_out, int out_size, void* d_ws,
                              size_t ws_size, hipStream_t stream) {
    const float*         x    = (const float*)d_in[0];
    const unsigned char* msk  = (const unsigned char*)d_in[1];
    const float* ln_g = (const float*)d_in[2];
    const float* ln_b = (const float*)d_in[3];
    const float* w1   = (const float*)d_in[4];
    const float* b1   = (const float*)d_in[5];
    const float* w2   = (const float*)d_in[6];
    const float* b2   = (const float*)d_in[7];
    const float* rot  = (const float*)d_in[8];
    const float* th0  = (const float*)d_in[9];
    const float* wh0  = (const float*)d_in[10];
    const float* wt0  = (const float*)d_in[11];
    const float* bt0  = (const float*)d_in[12];
    const float* th1  = (const float*)d_in[13];
    const float* wh1  = (const float*)d_in[14];
    const float* wt1  = (const float*)d_in[15];
    const float* bt1  = (const float*)d_in[16];
    float* out = (float*)d_out;

    unsigned char* wsb = (unsigned char*)d_ws;
    size_t off = 0;
    auto alloc = [&](size_t bytes) -> void* {
        off = (off + 255) & ~(size_t)255;
        void* p = wsb + off;
        off += bytes;
        return p;
    };

    unsigned short* xn_bf = (unsigned short*)alloc((size_t)Pp * Ff * 2);
    unsigned short* h1_bf = (unsigned short*)alloc((size_t)Pp * Dd * 2);
    float*          xd_f  = (float*)alloc((size_t)Pp * Dd * 4);
    unsigned short* xd_bf = (unsigned short*)alloc((size_t)Pp * Dd * 2);
    int* key  = (int*)alloc((size_t)Pp * 4);
    int* flat = (int*)alloc((size_t)Pp * 4);
    int* bcnt = (int*)alloc((size_t)(Pp / 256) * NKEY * 4);
    int* boff = (int*)alloc((size_t)(Pp / 256) * NKEY * 4);
    unsigned short* w1T  = (unsigned short*)alloc((size_t)Ff * Dd * 2);
    unsigned short* w2T  = (unsigned short*)alloc((size_t)Dd * Dd * 2);
    unsigned short* th0T = (unsigned short*)alloc((size_t)Ff * Ff * 2);
    unsigned short* wh0T = (unsigned short*)alloc((size_t)Ff * Ff * 2);
    unsigned short* wt0T = (unsigned short*)alloc((size_t)Ff * Ff * 2);
    unsigned short* th1T = (unsigned short*)alloc((size_t)Ff * Ff * 2);
    unsigned short* wh1T = (unsigned short*)alloc((size_t)Ff * Ff * 2);
    unsigned short* wt1T = (unsigned short*)alloc((size_t)Ff * Ff * 2);

    auto tr = [&](const float* w, unsigned short* wt_, int rows, int cols) {
        int n = rows * cols;
        k_transpose_bf16<<<(n + 255) / 256, 256, 0, stream>>>(w, wt_, rows, cols);
    };
    tr(w1, w1T, Ff, Dd);
    tr(w2, w2T, Dd, Dd);
    tr(th0, th0T, Ff, Ff); tr(wh0, wh0T, Ff, Ff); tr(wt0, wt0T, Ff, Ff);
    tr(th1, th1T, Ff, Ff); tr(wh1, wh1T, Ff, Ff); tr(wt1, wt1T, Ff, Ff);

    // LayerNorm: 8 waves/block, 1 wave/point
    k_layernorm<<<Pp / 8, 256, 0, stream>>>(x, ln_g, ln_b, xn_bf);

    // ffn_dist: elu(xn@w1+b1) then @w2+b2  (TDM-staged LDS tiles)
    size_t smem1 = ((size_t)128 * Ff + (size_t)Dd * Ff) * 2;  // A + BT, K=256
    size_t smem2 = ((size_t)128 * Dd + (size_t)Dd * Dd) * 2;  // A + BT, K=128
    k_gemm_bias_act<<<Pp / 128, 256, smem1, stream>>>(xn_bf, w1T, b1, h1_bf,
                                                      (float*)nullptr, Ff, Dd, 1);
    k_gemm_bias_act<<<Pp / 128, 256, smem2, stream>>>(h1_bf, w2T, b2, xd_bf,
                                                      xd_f, Dd, Dd, 0);

    // LSH bin assignment
    k_bins<<<Pp / 128, 128, 0, stream>>>(xd_f, rot, msk, key);

    // Deterministic stable counting sort -> flat permutation
    k_histb<<<Pp / 256, 256, 0, stream>>>(key, bcnt);
    k_scan<<<Bb, 32, 0, stream>>>(bcnt, boff);
    k_place<<<Pp / 256, 256, 0, stream>>>(key, boff, flat);

    // Fused per-bin Gram/adjacency + 2x GHConv, scatter to output
    size_t smem = (size_t)(BS * Dd + BS * BS + BS * Ff + Ff * BS + BS * Ff) * 2 +
                  (size_t)BS * 3 * 4 + (size_t)BS * 4;
    k_binproc<<<Bb * NBINS, 256, smem, stream>>>(
        xn_bf, xd_bf, xd_f, flat, msk, th0T, wh0T, wt0T, bt0, th1T, wh1T, wt1T,
        bt1, out);
}
